// mixed_decayRNN_fix_ib4_72464688218248
// MI455X (gfx1250) — matrix operand order repaired and checked by reading, not compiled, not run on hardware
//
#include <hip/hip_runtime.h>

typedef __attribute__((ext_vector_type(16))) __bf16 v16bf;
typedef __attribute__((ext_vector_type(8)))  __bf16 v8bf;
typedef __attribute__((ext_vector_type(8)))  float  v8f;

namespace {
constexpr int Bn = 256;   // batch
constexpr int Tn = 256;   // time
constexpr int In = 5;     // input dim
constexpr int Hn = 2048;  // hidden dim
constexpr int On = 4;     // output dim
constexpr float kAlpha = 0.1f;
}

// ---------------------------------------------------------------------------
// One-time prep: Wrec f32 -> bf16 (row-major [H][H], row j = B-matrix column j
// of Wrec^T), and zero the initial hidden-state double buffers.
// ---------------------------------------------------------------------------
__global__ __launch_bounds__(256) void init_state_kernel(
    const float* __restrict__ Wrec, __bf16* __restrict__ Wrec_bf,
    float* __restrict__ h_f32, __bf16* __restrict__ h_bf) {
  int idx = blockIdx.x * 256 + threadIdx.x;
  if (idx < Hn * Hn) Wrec_bf[idx] = (__bf16)Wrec[idx];
  if (idx < Bn * Hn) {
    h_f32[idx] = 0.0f;
    h_bf[idx]  = (__bf16)0.0f;
  }
}

union V16 { v16bf v; v8bf h[2]; };

// ---------------------------------------------------------------------------
// One recurrent step:
//   pre[b,j] = (h_bf16 @ Wrec_bf16^T)[b,j] + x[b,t,:]·Win[j,:] + noise[t,b,j]
//   h_new    = relu(pre)*alpha + (1-alpha)*h_f32_old
// Writes h_new into the *other* state buffer (double-buffered) and into the
// hiddens output [B,T,H].
//
// Block = 128 threads = 4 waves arranged 2x2; wave tile = 32(M) x 32(N) via
// 2x2 v_wmma_f32_16x16x32_bf16 accumulators. Grid = (H/64, B/64) = (32, 4).
// ---------------------------------------------------------------------------
__global__ __launch_bounds__(128) void rnn_step_kernel(
    const float* __restrict__ x,         // [B,T,I]
    const float* __restrict__ Win,       // [H,I]
    const __bf16* __restrict__ Wrec_bf,  // [H,H] row-major: row j holds Wrec[j,k]
    const float* __restrict__ noise,     // [T,B,H]
    const float* __restrict__ h_in_f32,  // [B,H]
    const __bf16* __restrict__ h_in_bf,  // [B,H]
    float* __restrict__ h_out_f32,       // [B,H]
    __bf16* __restrict__ h_out_bf,       // [B,H]
    float* __restrict__ hiddens,         // [B,T,H]
    int t) {
  const int lane = threadIdx.x & 31;
  const int wave = threadIdx.x >> 5;
  const int m0 = blockIdx.y * 64 + (wave >> 1) * 32;  // batch rows
  const int n0 = blockIdx.x * 64 + (wave & 1) * 32;   // hidden cols
  const int half = lane >> 4;
  const int l16 = lane & 15;

  // A fragment base pointers (h rows), per documented 16-bit 16x32 layout:
  // lane holds row (l&15); elements 0..7 = K base..base+7, 8..15 = K base+16..+23,
  // base = (lane>>4)*8  -> two contiguous 16B chunks.
  const __bf16* a0p = h_in_bf + (size_t)(m0 + l16) * Hn + half * 8;
  const __bf16* a1p = a0p + (size_t)16 * Hn;
  // B fragment base pointers (Wrec rows = columns of Wrec^T): lane holds
  // column (l&15); same K order as A: base..base+7, base+16..+23, base=(lane>>4)*8.
  const __bf16* b0p = Wrec_bf + (size_t)(n0 + l16) * Hn + half * 8;
  const __bf16* b1p = b0p + (size_t)16 * Hn;

  v8f acc00 = {}, acc01 = {}, acc10 = {}, acc11 = {};

  for (int k0 = 0; k0 < Hn; k0 += 32) {
    // prefetch next K-chunk toward near cache (data is L2-resident)
    __builtin_prefetch(b0p + k0 + 32, 0, 3);
    __builtin_prefetch(b1p + k0 + 32, 0, 3);
    V16 a0, a1, b0, b1;
    a0.h[0] = *(const v8bf*)(a0p + k0);
    a0.h[1] = *(const v8bf*)(a0p + k0 + 16);
    a1.h[0] = *(const v8bf*)(a1p + k0);
    a1.h[1] = *(const v8bf*)(a1p + k0 + 16);
    b0.h[0] = *(const v8bf*)(b0p + k0);
    b0.h[1] = *(const v8bf*)(b0p + k0 + 16);
    b1.h[0] = *(const v8bf*)(b1p + k0);
    b1.h[1] = *(const v8bf*)(b1p + k0 + 16);
    acc00 = __builtin_amdgcn_wmma_f32_16x16x32_bf16(false, a0.v, false, b0.v,
                                                    (short)0, acc00, false, false);
    acc01 = __builtin_amdgcn_wmma_f32_16x16x32_bf16(false, a0.v, false, b1.v,
                                                    (short)0, acc01, false, false);
    acc10 = __builtin_amdgcn_wmma_f32_16x16x32_bf16(false, a1.v, false, b0.v,
                                                    (short)0, acc10, false, false);
    acc11 = __builtin_amdgcn_wmma_f32_16x16x32_bf16(false, a1.v, false, b1.v,
                                                    (short)0, acc11, false, false);
  }

  // Epilogue: C/D layout -> lane holds column n = nbase + (l&15),
  // VGPR j holds row m = mbase + j + 8*(lane>>4).
  auto epilogue = [&](const v8f& acc, int mbase, int nbase) {
    const int n = nbase + l16;
    const float w0 = Win[n * In + 0];
    const float w1 = Win[n * In + 1];
    const float w2 = Win[n * In + 2];
    const float w3 = Win[n * In + 3];
    const float w4 = Win[n * In + 4];
#pragma unroll
    for (int j = 0; j < 8; ++j) {
      const int m = mbase + j + 8 * half;
      const float* xr = x + ((size_t)m * Tn + t) * In;
      const float it = xr[0] * w0 + xr[1] * w1 + xr[2] * w2 + xr[3] * w3 + xr[4] * w4;
      const float nt = noise[((size_t)t * Bn + m) * Hn + n];
      const float ho = h_in_f32[(size_t)m * Hn + n];
      const float pre = acc[j] + it + nt;
      const float relu = pre > 0.0f ? pre : 0.0f;
      const float hn = relu * kAlpha + (1.0f - kAlpha) * ho;
      h_out_f32[(size_t)m * Hn + n] = hn;
      h_out_bf[(size_t)m * Hn + n] = (__bf16)hn;
      hiddens[((size_t)m * Tn + t) * Hn + n] = hn;
    }
  };
  epilogue(acc00, m0, n0);
  epilogue(acc01, m0, n0 + 16);
  epilogue(acc10, m0 + 16, n0);
  epilogue(acc11, m0 + 16, n0 + 16);
}

// ---------------------------------------------------------------------------
// Output head per step: logits[b,o] = h_new[b,:]·Wout[o,:], softmax over O=4.
// One block per batch row; Wout (32KB) is L2-resident.
// ---------------------------------------------------------------------------
__global__ __launch_bounds__(256) void head_kernel(
    const float* __restrict__ h,     // [B,H] (fresh h_new)
    const float* __restrict__ Wout,  // [O,H]
    float* __restrict__ outputs,     // [B,T,O]
    int t) {
  const int b = blockIdx.x;
  const int tid = threadIdx.x;
  const float* hr = h + (size_t)b * Hn;

  float a0 = 0.f, a1 = 0.f, a2 = 0.f, a3 = 0.f;
  for (int j = tid; j < Hn; j += 256) {
    const float hv = hr[j];
    a0 += hv * Wout[0 * Hn + j];
    a1 += hv * Wout[1 * Hn + j];
    a2 += hv * Wout[2 * Hn + j];
    a3 += hv * Wout[3 * Hn + j];
  }
  __shared__ float red[4][256];
  red[0][tid] = a0; red[1][tid] = a1; red[2][tid] = a2; red[3][tid] = a3;
  __syncthreads();
  for (int s = 128; s > 0; s >>= 1) {
    if (tid < s) {
      red[0][tid] += red[0][tid + s];
      red[1][tid] += red[1][tid + s];
      red[2][tid] += red[2][tid + s];
      red[3][tid] += red[3][tid + s];
    }
    __syncthreads();
  }
  if (tid == 0) {
    const float l0 = red[0][0], l1 = red[1][0], l2 = red[2][0], l3 = red[3][0];
    const float mx = fmaxf(fmaxf(l0, l1), fmaxf(l2, l3));
    const float e0 = expf(l0 - mx), e1 = expf(l1 - mx);
    const float e2 = expf(l2 - mx), e3 = expf(l3 - mx);
    const float inv = 1.0f / (e0 + e1 + e2 + e3);
    float* op = outputs + ((size_t)b * Tn + t) * On;
    op[0] = e0 * inv; op[1] = e1 * inv; op[2] = e2 * inv; op[3] = e3 * inv;
  }
}

// ---------------------------------------------------------------------------
extern "C" void kernel_launch(void* const* d_in, const int* in_sizes, int n_in,
                              void* d_out, int out_size, void* d_ws, size_t ws_size,
                              hipStream_t stream) {
  (void)in_sizes; (void)n_in; (void)out_size; (void)ws_size;
  const float* x     = (const float*)d_in[0];  // [B,T,I]
  const float* Win   = (const float*)d_in[1];  // [H,I]
  const float* Wrec  = (const float*)d_in[2];  // [H,H]
  const float* Wout  = (const float*)d_in[3];  // [O,H]
  const float* noise = (const float*)d_in[4];  // [T,B,H]

  float* hiddens = (float*)d_out;                       // [B,T,H]
  float* outputs = hiddens + (size_t)Bn * Tn * Hn;      // [B,T,O]

  // Workspace layout (double-buffered hidden state + bf16 Wrec): ~14 MB.
  char* ws = (char*)d_ws;
  __bf16* Wrec_bf = (__bf16*)ws;            ws += (size_t)Hn * Hn * sizeof(__bf16);
  float* hf0 = (float*)ws;                  ws += (size_t)Bn * Hn * sizeof(float);
  float* hf1 = (float*)ws;                  ws += (size_t)Bn * Hn * sizeof(float);
  __bf16* hb0 = (__bf16*)ws;               ws += (size_t)Bn * Hn * sizeof(__bf16);
  __bf16* hb1 = (__bf16*)ws;
  float* hf[2] = {hf0, hf1};
  __bf16* hb[2] = {hb0, hb1};

  init_state_kernel<<<(Hn * Hn + 255) / 256, 256, 0, stream>>>(Wrec, Wrec_bf, hf[0], hb[0]);

  const dim3 grid(Hn / 64, Bn / 64);  // (32, 4)
  for (int t = 0; t < Tn; ++t) {
    const int cur = t & 1;
    const int nxt = cur ^ 1;
    rnn_step_kernel<<<grid, 128, 0, stream>>>(x, Win, Wrec_bf, noise,
                                              hf[cur], hb[cur], hf[nxt], hb[nxt],
                                              hiddens, t);
    head_kernel<<<Bn, 256, 0, stream>>>(hf[nxt], Wout, outputs, t);
  }
}
